// CNNT_enhanced_denoising_runtime_53704271069472
// MI455X (gfx1250) — matrix operand order repaired and checked by reading, not compiled, not run on hardware
//
#include <hip/hip_runtime.h>

// ---------------------------------------------------------------------------
// MI455X (gfx1250) implementation. Roofline: ~47 GFLOP, ~1.1 GB HBM traffic
// @ 23.3 TB/s => ~50us memory floor. bf16 WMMA (v_wmma_f32_16x16x32_bf16,
// f32 accumulate) keeps every GEMM stage (3x3 convs as implicit GEMM
// M=pixels,N=16,K=144; QK^T M=N=64,K=32768; PV M=64,K=64,N=32768) far below
// the memory roofline. Intermediates bf16 to halve traffic.
//
// Conv K-dim ordered kk = (ky*3+kx)*16 + c so every A-fragment is 2x
// ds_load_b128 at immediate offsets. V staged transposed in LDS so PV
// B-frags are contiguous. Q/K/P staging uses GLOBAL_LOAD_ASYNC_TO_LDS_B128
// (+ s_wait_asynccnt) -- builtin confirmed present; casts now match its
// exact parameter types (int4 vector ptr, AS1 src / AS3 LDS dst).
// ---------------------------------------------------------------------------

#define NT 256

typedef __attribute__((ext_vector_type(16))) __bf16   v16bf;
typedef __attribute__((ext_vector_type(8)))  float    v8f;
typedef __attribute__((ext_vector_type(4)))  unsigned uint4v;
typedef __attribute__((ext_vector_type(4)))  int      int4v;
typedef __attribute__((ext_vector_type(4)))  float    float4v;

union FragAB {
  v16bf v;
  unsigned short h[16];
  uint4v u4[2];
};

union U8 {
  uint4v u;
  unsigned short h[8];
};

__device__ __forceinline__ unsigned short f2bf(float f) {
  unsigned u = __float_as_uint(f);
  u += 0x7FFFu + ((u >> 16) & 1u);          // round-to-nearest-even
  return (unsigned short)(u >> 16);
}

__device__ __forceinline__ v8f wmma_bf(const FragAB& a, const FragAB& b, v8f c) {
  // emits v_wmma_f32_16x16x32_bf16
  return __builtin_amdgcn_wmma_f32_16x16x32_bf16(false, a.v, false, b.v,
                                                 (short)0, c, false, false);
}

// --- CDNA5 async global->LDS copy (ASYNCcnt path), with safe fallback ------
#if defined(__has_builtin)
#if __has_builtin(__builtin_amdgcn_global_load_async_to_lds_b128) && \
    __has_builtin(__builtin_amdgcn_s_wait_asynccnt)
#define HAVE_ASYNC_LDS 1
#endif
#endif

#ifdef HAVE_ASYNC_LDS
typedef __attribute__((address_space(1))) int4v gint4;   // prints as "__device__"
typedef __attribute__((address_space(3))) int4v lint4;   // LDS
#endif

__device__ __forceinline__ void async_copy_b128(const unsigned short* g,
                                                unsigned short* l) {
#ifdef HAVE_ASYNC_LDS
  // generic LDS pointer: [31:0] is the LDS offset (aperture rules)
  __builtin_amdgcn_global_load_async_to_lds_b128(
      (gint4*)(unsigned long long)g,
      (lint4*)(unsigned)(unsigned long long)l, 0, 0);
#else
  *(uint4v*)l = *(const uint4v*)g;
#endif
}

__device__ __forceinline__ void async_wait() {
#ifdef HAVE_ASYNC_LDS
  __builtin_amdgcn_s_wait_asynccnt(0);
#endif
}

// ---------------------------------------------------------------------------
// Kernel 1: pack conv weights into B-frag layout for 16x16x32 bf16 WMMA.
// K ordering: kk = tap*16 + c, tap = ky*3 + kx (so tap is compile-time in the
// conv gathers and c is LDS-contiguous). lane = half*16+n, element i ->
// K = step*32 + (i>>3)*16 + half*8 + (i&7). convs: 0=wq 1=wk 2=wv 3=wo; 144 padded to 160.
// ---------------------------------------------------------------------------
__global__ void pack_w_kernel(const float* __restrict__ wq,
                              const float* __restrict__ wk,
                              const float* __restrict__ wv,
                              const float* __restrict__ wo,
                              unsigned short* __restrict__ wpack) {
  int idx = blockIdx.x * NT + threadIdx.x;
  if (idx >= 4 * 5 * 32 * 16) return;
  int i    = idx & 15;
  int lane = (idx >> 4) & 31;
  int cs   = idx >> 9;            // conv*5 + step
  int step = cs % 5;
  int conv = cs / 5;
  int n    = lane & 15;
  int half = lane >> 4;
  int kk   = step * 32 + (i >> 3) * 16 + half * 8 + (i & 7);
  int tap  = kk >> 4;             // ky*3 + kx
  int c    = kk & 15;
  const float* w = (conv == 0) ? wq : (conv == 1) ? wk : (conv == 2) ? wv : wo;
  unsigned short val = 0;
  if (tap < 9) val = f2bf(w[n * 144 + c * 9 + tap]);   // w[o][c][ky][kx]
  wpack[idx] = val;
}

// ---------------------------------------------------------------------------
// Kernel 2: fused q/k/v 3x3 conv as implicit GEMM. One block per (image bt,
// output row yy). LDS: 3-row x 130-col x 16-ch bf16 halo (c fastest). Each of
// 8 waves produces a 16-pixel x 16-outch tile; A-frag = 2x b128 LDS loads at
// compile-time tap offsets, reused across the three convs.
// Output: q/k/v[b][h][t][d], d = c_local*16384 + yy*128 + x, bf16.
// ---------------------------------------------------------------------------
__global__ void conv_qkv_kernel(const float* __restrict__ x,
                                const float* __restrict__ bq,
                                const float* __restrict__ bk,
                                const float* __restrict__ bv,
                                const unsigned short* __restrict__ wpack,
                                unsigned short* __restrict__ qo,
                                unsigned short* __restrict__ ko,
                                unsigned short* __restrict__ vo) {
  const int bt = blockIdx.x >> 7;
  const int yy = blockIdx.x & 127;
  __shared__ __align__(16) unsigned short in_s[3 * 130 * 16];

  for (int idx = threadIdx.x; idx < 3 * 130 * 16; idx += NT) {
    int c   = idx & 15;
    int col = (idx >> 4) % 130;
    int ky  = idx / (130 * 16);
    int ys  = yy - 1 + ky;
    int xs  = col - 1;
    float val = 0.f;
    if (ys >= 0 && ys < 128 && xs >= 0 && xs < 128)
      val = x[((bt * 16 + c) << 14) + (ys << 7) + xs];
    in_s[idx] = f2bf(val);
  }
  __syncthreads();

  const int lane = threadIdx.x & 31;
  const int wid  = threadIdx.x >> 5;
  const int half = lane >> 4;
  const int mrow = lane & 15;
  const int x0   = wid << 4;
  const int rb   = (x0 + mrow) * 16 + half * 8;  // runtime base for A gathers

  v8f z = {0.f, 0.f, 0.f, 0.f, 0.f, 0.f, 0.f, 0.f};
  v8f acc[3] = {z, z, z};

#pragma unroll
  for (int step = 0; step < 5; ++step) {
    FragAB a;
    {
      const int tap0 = step * 2;
      const int c0 = ((tap0 / 3) * 130 + (tap0 % 3)) * 16;  // compile-time
      a.u4[0] = *(const uint4v*)&in_s[rb + c0];
    }
    {
      const int tap1 = step * 2 + 1;
      if (tap1 < 9) {                                        // compile-time
        const int c1 = ((tap1 / 3) * 130 + (tap1 % 3)) * 16;
        a.u4[1] = *(const uint4v*)&in_s[rb + c1];
      } else {
        uint4v zz = {0u, 0u, 0u, 0u};
        a.u4[1] = zz;                                        // K padding
      }
    }
#pragma unroll
    for (int cv = 0; cv < 3; ++cv) {
      FragAB bf;
      const uint4v* wp =
          (const uint4v*)(wpack + (((cv * 5 + step) * 32 + lane) << 4));
      bf.u4[0] = wp[0];
      bf.u4[1] = wp[1];
      acc[cv] = wmma_bf(a, bf, acc[cv]);
    }
  }

  const int o = mrow;                 // C/D: N = lane%16
  const int b = bt >> 6, t = bt & 63;
  const int h = o >> 1, cl = o & 1;
  const int base = (((b * 8 + h) * 64 + t) << 15) + (cl << 14) + (yy << 7) +
                   x0 + half * 8;     // M = r + half*8 -> consecutive x
  float bias[3] = {bq[o], bk[o], bv[o]};
  unsigned short* outs[3] = {qo, ko, vo};
#pragma unroll
  for (int cv = 0; cv < 3; ++cv) {
    uint4v pk;
#pragma unroll
    for (int j = 0; j < 4; ++j) {
      unsigned lo = f2bf(acc[cv][2 * j]     + bias[cv]);
      unsigned hi = f2bf(acc[cv][2 * j + 1] + bias[cv]);
      pk[j] = lo | (hi << 16);
    }
    *(uint4v*)(outs[cv] + base) = pk;
  }
}

// ---------------------------------------------------------------------------
// Kernel 3: partial attention scores. S = Q K^T, M=N=64, K=32768 per (b,h).
// Grid = 16 (b,h) x 16 K-slices (2048 each); partials written per-slice
// (deterministic reduction in softmax). 64-wide D chunks of Q and K staged in
// LDS via async b128 copies; 8 waves x 2 tiles cover the 64x64 output.
// ---------------------------------------------------------------------------
__global__ void attn_scores_kernel(const unsigned short* __restrict__ qb,
                                   const unsigned short* __restrict__ kb,
                                   float* __restrict__ spart) {
  const int bh = blockIdx.x >> 4;
  const int ks = blockIdx.x & 15;
  __shared__ __align__(16) unsigned short Qs[4096];
  __shared__ __align__(16) unsigned short Ks[4096];

  const int lane = threadIdx.x & 31;
  const int wid  = threadIdx.x >> 5;
  const int half = lane >> 4;
  const int mrow = lane & 15;
  const int Mt0 = wid >> 2, Nt = wid & 3;
  const int Mt1 = (wid + 8) >> 2;             // second tile: same Nt, Mt+2

  v8f z = {0.f, 0.f, 0.f, 0.f, 0.f, 0.f, 0.f, 0.f};
  v8f acc0 = z, acc1 = z;
  const int rowbase = bh * 64 * 32768 + ks * 2048;

  for (int ch = 0; ch < 32; ++ch) {
    for (int e = threadIdx.x; e < 512; e += NT) {
      int off = e << 3;
      int tt = off >> 6, dl = off & 63;
      int g = rowbase + tt * 32768 + ch * 64 + dl;
      async_copy_b128(qb + g, &Qs[off]);
      async_copy_b128(kb + g, &Ks[off]);
    }
    async_wait();
    __syncthreads();
#pragma unroll
    for (int kst = 0; kst < 2; ++kst) {
      FragAB a0, a1, bb;
      const int abase = kst * 32 + half * 8;
      a0.u4[0] = *(const uint4v*)&Qs[((Mt0 * 16 + mrow) << 6) + abase];
      a0.u4[1] = *(const uint4v*)&Qs[((Mt0 * 16 + mrow) << 6) + abase + 16];
      a1.u4[0] = *(const uint4v*)&Qs[((Mt1 * 16 + mrow) << 6) + abase];
      a1.u4[1] = *(const uint4v*)&Qs[((Mt1 * 16 + mrow) << 6) + abase + 16];
      const int bbase = kst * 32 + half * 8;
      bb.u4[0] = *(const uint4v*)&Ks[((Nt * 16 + mrow) << 6) + bbase];
      bb.u4[1] = *(const uint4v*)&Ks[((Nt * 16 + mrow) << 6) + bbase + 16];
      acc0 = wmma_bf(a0, bb, acc0);
      acc1 = wmma_bf(a1, bb, acc1);
    }
    __syncthreads();
  }

  const int s = Nt * 16 + mrow;
#pragma unroll
  for (int r = 0; r < 8; ++r) {
    int tA = Mt0 * 16 + half * 8 + r;
    int tB = Mt1 * 16 + half * 8 + r;
    spart[(((ks * 16 + bh) * 64 + tA) << 6) + s] = acc0[r];
    spart[(((ks * 16 + bh) * 64 + tB) << 6) + s] = acc1[r];
  }
}

// ---------------------------------------------------------------------------
// Kernel 4: reduce the 16 K-slice partials, scale by 1/sqrt(32768), causal
// softmax, write P as bf16. 16 blocks x 64 threads (one row per thread).
// ---------------------------------------------------------------------------
__global__ void softmax_kernel(const float* __restrict__ spart,
                               unsigned short* __restrict__ P) {
  const int bh = blockIdx.x;
  const int t  = threadIdx.x;
  const float scale = 0.00552427172801990326f;  // 1/sqrt(32768)
  float row[64];
  for (int s = 0; s < 64; ++s) {
    float v = 0.f;
    for (int ks = 0; ks < 16; ++ks)
      v += spart[(((ks * 16 + bh) * 64 + t) << 6) + s];
    row[s] = v * scale;
  }
  float m = -3.402823466e38f;
  for (int s = 0; s <= t; ++s) m = fmaxf(m, row[s]);
  float sum = 0.f;
  for (int s = 0; s <= t; ++s) {
    float e = __expf(row[s] - m);
    row[s] = e;
    sum += e;
  }
  float inv = 1.f / sum;
  for (int s = 0; s < 64; ++s) {
    float p = (s <= t) ? row[s] * inv : 0.f;
    P[((bh * 64 + t) << 6) + s] = f2bf(p);
  }
}

// ---------------------------------------------------------------------------
// Kernel 5: Y = P V. Grid over (b,h) x 512 chunks of 64 along D. P async-
// staged row-major; the V chunk staged TRANSPOSED (Vt[d][s]) so B-fragments
// are contiguous b128 LDS loads. 2 WMMA K-steps. Output written in
// [b][t][o][yy][xx] bf16 layout ready for the final conv.
// ---------------------------------------------------------------------------
__global__ void attn_pv_kernel(const unsigned short* __restrict__ P,
                               const unsigned short* __restrict__ vb,
                               unsigned short* __restrict__ yws) {
  const int bh = blockIdx.x >> 9;
  const int dc = blockIdx.x & 511;
  __shared__ __align__(16) unsigned short Ps[4096];
  __shared__ __align__(16) unsigned short Vt[4096];   // [d][s]

  for (int e = threadIdx.x; e < 512; e += NT) {
    int off = e << 3;
    async_copy_b128(P + (bh << 12) + off, &Ps[off]);
    int s = off >> 6, dl = off & 63;
    U8 tmp;
    tmp.u = *(const uint4v*)(vb + ((bh * 64 + s) << 15) + dc * 64 + dl);
#pragma unroll
    for (int j = 0; j < 8; ++j) Vt[(dl + j) * 64 + s] = tmp.h[j];
  }
  async_wait();
  __syncthreads();

  const int lane = threadIdx.x & 31;
  const int wid  = threadIdx.x >> 5;
  const int half = lane >> 4;
  const int mrow = lane & 15;
  const int Mt0 = wid >> 2, Nt = wid & 3;
  const int Mt1 = (wid + 8) >> 2;

  v8f z = {0.f, 0.f, 0.f, 0.f, 0.f, 0.f, 0.f, 0.f};
  v8f acc0 = z, acc1 = z;
#pragma unroll
  for (int kst = 0; kst < 2; ++kst) {
    FragAB a0, a1, bb;
    const int abase = kst * 32 + half * 8;
    a0.u4[0] = *(const uint4v*)&Ps[((Mt0 * 16 + mrow) << 6) + abase];
    a0.u4[1] = *(const uint4v*)&Ps[((Mt0 * 16 + mrow) << 6) + abase + 16];
    a1.u4[0] = *(const uint4v*)&Ps[((Mt1 * 16 + mrow) << 6) + abase];
    a1.u4[1] = *(const uint4v*)&Ps[((Mt1 * 16 + mrow) << 6) + abase + 16];
    const int bbase = kst * 32 + half * 8;            // s index
    bb.u4[0] = *(const uint4v*)&Vt[((Nt * 16 + mrow) << 6) + bbase];
    bb.u4[1] = *(const uint4v*)&Vt[((Nt * 16 + mrow) << 6) + bbase + 16];
    acc0 = wmma_bf(a0, bb, acc0);
    acc1 = wmma_bf(a1, bb, acc1);
  }

  const int b = bh >> 3, h = bh & 7;
  const int dglob = dc * 64 + Nt * 16 + mrow;
  const int cl = dglob >> 14, rem = dglob & 16383;
  const int o = h * 2 + cl;
#pragma unroll
  for (int r = 0; r < 8; ++r) {
    int tA = Mt0 * 16 + half * 8 + r;
    int tB = Mt1 * 16 + half * 8 + r;
    yws[(((b * 64 + tA) * 16 + o) << 14) + rem] = f2bf(acc0[r]);
    yws[(((b * 64 + tB) * 16 + o) << 14) + rem] = f2bf(acc1[r]);
  }
}

// ---------------------------------------------------------------------------
// Kernel 6: output 3x3 conv (bf16 in, f32 out to d_out). Same implicit-GEMM
// structure as conv_qkv, single accumulator, conv index 3 in wpack.
// ---------------------------------------------------------------------------
__global__ void conv_out_kernel(const unsigned short* __restrict__ yws,
                                const float* __restrict__ bo,
                                const unsigned short* __restrict__ wpack,
                                float* __restrict__ out) {
  const int bt = blockIdx.x >> 7;
  const int yy = blockIdx.x & 127;
  __shared__ __align__(16) unsigned short in_s[3 * 130 * 16];

  for (int idx = threadIdx.x; idx < 3 * 130 * 16; idx += NT) {
    int c   = idx & 15;
    int col = (idx >> 4) % 130;
    int ky  = idx / (130 * 16);
    int ys  = yy - 1 + ky;
    int xs  = col - 1;
    unsigned short val = 0;
    if (ys >= 0 && ys < 128 && xs >= 0 && xs < 128)
      val = yws[((bt * 16 + c) << 14) + (ys << 7) + xs];
    in_s[idx] = val;
  }
  __syncthreads();

  const int lane = threadIdx.x & 31;
  const int wid  = threadIdx.x >> 5;
  const int half = lane >> 4;
  const int mrow = lane & 15;
  const int x0   = wid << 4;
  const int rb   = (x0 + mrow) * 16 + half * 8;

  v8f acc = {0.f, 0.f, 0.f, 0.f, 0.f, 0.f, 0.f, 0.f};
#pragma unroll
  for (int step = 0; step < 5; ++step) {
    FragAB a;
    {
      const int tap0 = step * 2;
      const int c0 = ((tap0 / 3) * 130 + (tap0 % 3)) * 16;
      a.u4[0] = *(const uint4v*)&in_s[rb + c0];
    }
    {
      const int tap1 = step * 2 + 1;
      if (tap1 < 9) {
        const int c1 = ((tap1 / 3) * 130 + (tap1 % 3)) * 16;
        a.u4[1] = *(const uint4v*)&in_s[rb + c1];
      } else {
        uint4v zz = {0u, 0u, 0u, 0u};
        a.u4[1] = zz;
      }
    }
    FragAB bf;
    const uint4v* wp =
        (const uint4v*)(wpack + (((3 * 5 + step) * 32 + lane) << 4));
    bf.u4[0] = wp[0];
    bf.u4[1] = wp[1];
    acc = wmma_bf(a, bf, acc);
  }

  const int o = mrow;
  const float bias = bo[o];
  const int base = ((bt * 16 + o) << 14) + (yy << 7) + x0 + half * 8;
  float4v f0, f1;
#pragma unroll
  for (int j = 0; j < 4; ++j) {
    f0[j] = acc[j] + bias;
    f1[j] = acc[4 + j] + bias;
  }
  *(float4v*)(out + base)     = f0;
  *(float4v*)(out + base + 4) = f1;
}

// ---------------------------------------------------------------------------
extern "C" void kernel_launch(void* const* d_in, const int* in_sizes, int n_in,
                              void* d_out, int out_size, void* d_ws, size_t ws_size,
                              hipStream_t stream) {
  (void)in_sizes; (void)n_in; (void)out_size; (void)ws_size;
  const float* x  = (const float*)d_in[0];
  const float* wq = (const float*)d_in[1];
  const float* bq = (const float*)d_in[2];
  const float* wk = (const float*)d_in[3];
  const float* bk = (const float*)d_in[4];
  const float* wv = (const float*)d_in[5];
  const float* bv = (const float*)d_in[6];
  const float* wo = (const float*)d_in[7];
  const float* bo = (const float*)d_in[8];
  float* out = (float*)d_out;

  // Workspace (~196 MB): q|k|v bf16 (64MB each), S partials f32 (4MB),
  // P bf16 (128KB), packed weights. y intermediate aliases q (dead after QK^T).
  const size_t QKV = 33554432;  // elements per q/k/v
  unsigned short* qb = (unsigned short*)d_ws;
  unsigned short* kb = qb + QKV;
  unsigned short* vb = kb + QKV;
  float* spart       = (float*)(vb + QKV);
  unsigned short* P  = (unsigned short*)(spart + 16 * 16 * 64 * 64);
  unsigned short* wpack = P + 16 * 64 * 64;
  unsigned short* yws = qb;  // alias: q not read after attn_scores_kernel

  pack_w_kernel     <<<40,    NT, 0, stream>>>(wq, wk, wv, wo, wpack);
  conv_qkv_kernel   <<<16384, NT, 0, stream>>>(x, bq, bk, bv, wpack, qb, kb, vb);
  attn_scores_kernel<<<256,   NT, 0, stream>>>(qb, kb, spart);
  softmax_kernel    <<<16,    64, 0, stream>>>(spart, P);
  attn_pv_kernel    <<<8192,  NT, 0, stream>>>(P, vb, yws);
  conv_out_kernel   <<<16384, NT, 0, stream>>>(yws, bo, wpack, out);
}
